// MultiScaleWindowCrossAttention_71038759075977
// MI455X (gfx1250) — matrix operand order repaired and checked by reading, not compiled, not run on hardware
//
#include <hip/hip_runtime.h>

typedef __attribute__((ext_vector_type(16))) __bf16 v16bf;
typedef __attribute__((ext_vector_type(8)))  float  v8f;

#define Bv    8
#define C1    256
#define C2    128
#define H1c   128
#define W1c   128
#define NHEAD 8
#define HD    32
#define L1    256
#define L2    64
#define NWIN  512          // 8 * 64 windows (both partitions)

__device__ __forceinline__ v8f v8f_zero() {
  v8f z = {0.f, 0.f, 0.f, 0.f, 0.f, 0.f, 0.f, 0.f};
  return z;
}

__device__ __forceinline__ v8f wmma_bf16(v16bf a, v16bf b, v8f c) {
  // D = A(16x32 bf16) * B(32x16 bf16) + C(16x16 f32)
  return __builtin_amdgcn_wmma_f32_16x16x32_bf16(false, a, false, b,
                                                 (short)0, c, false, false);
}

// ---------------------------------------------------------------------------
// Prep: f32 -> bf16 bulk convert (used for the weight matrices, ~1.2 MB total)
// ---------------------------------------------------------------------------
__global__ __launch_bounds__(256) void k_cvt(const float* __restrict__ s,
                                             __bf16* __restrict__ d, int n) {
  const int i = blockIdx.x * 256 + threadIdx.x;
  if (i < n) d[i] = (__bf16)s[i];
}

// ---------------------------------------------------------------------------
// Kernel 1: mem_m = 1x1 conv (GEMM M=32768 tokens, N=256, K=128)
// mem is NCHW; outputs channels-last f32 (ghost branch) + bf16 (KV GEMM A)
// Each wave: 16 tokens x 128 channels (8 accumulators)
// ---------------------------------------------------------------------------
__global__ __launch_bounds__(256) void k_conv1x1(
    const float* __restrict__ mem, const __bf16* __restrict__ wb,
    const float* __restrict__ bias, float* __restrict__ mem_m,
    __bf16* __restrict__ mem_mb) {
  const int lane = threadIdx.x & 31, wv = threadIdx.x >> 5;
  const int half = lane >> 4, l16 = lane & 15;
  const int mtile = blockIdx.x * 8 + wv;
  const int n0 = blockIdx.y * 128;

  const int t = mtile * 16 + l16;                       // A row for this lane
  const int b = t >> 12, rem = t & 4095;
  const int y = rem >> 6, x = rem & 63;
  const float* abase = mem + ((size_t)b * C2) * 4096 + (size_t)y * 64 + x;

  v8f acc[8];
#pragma unroll
  for (int g = 0; g < 8; ++g) acc[g] = v8f_zero();

  for (int kk = 0; kk < C2; kk += 32) {
    v16bf a;
    const int kb = kk + half * 16;
#pragma unroll
    for (int j = 0; j < 16; ++j) {
      a[j]     = (__bf16)abase[(size_t)(kb + j) * 4096];
    }
#pragma unroll
    for (int g = 0; g < 8; ++g) {
      const int n = n0 + g * 16 + l16;
      const __bf16* wrow = wb + (size_t)n * C2 + kk + half * 16;
      v16bf bf;
#pragma unroll
      for (int j = 0; j < 16; ++j) bf[j] = wrow[j];
      acc[g] = wmma_bf16(a, bf, acc[g]);
    }
  }
#pragma unroll
  for (int g = 0; g < 8; ++g) {
    const int n = n0 + g * 16 + l16;
    const float bi = bias[n];
#pragma unroll
    for (int r = 0; r < 8; ++r) {
      const int tok = mtile * 16 + r + half * 8;
      const float v = acc[g][r] + bi;
      mem_m[(size_t)tok * C1 + n]  = v;
      mem_mb[(size_t)tok * C1 + n] = (__bf16)v;
    }
  }
}

// ---------------------------------------------------------------------------
// Kernel 2: q = linear(window(tgt)) + fused l2norm -> bf16
// qn layout: [win][head][tok(256)][c(32)]. 16 tokens x 128 channels per wave.
// ---------------------------------------------------------------------------
__global__ __launch_bounds__(256) void k_qnorm(
    const float* __restrict__ tgt, const __bf16* __restrict__ qwb,
    const float* __restrict__ qb, __bf16* __restrict__ qn) {
  const int lane = threadIdx.x & 31, wv = threadIdx.x >> 5;
  const int half = lane >> 4, l16 = lane & 15;
  const int mtile = blockIdx.x * 8 + wv;
  const int n0 = blockIdx.y * 128;

  // window-partition mapping for this lane's A row
  const int t = mtile * 16 + l16;
  const int win = t >> 8, li = t & 255;
  const int ii = li >> 4, jj = li & 15;
  const int b = win >> 6, wq = win & 63;
  const int y = (wq >> 3) * 16 + ii, x = (wq & 7) * 16 + jj;
  const float* abase = tgt + ((size_t)b * C1) * 16384 + (size_t)y * 128 + x;

  v8f acc[8];
#pragma unroll
  for (int g = 0; g < 8; ++g) acc[g] = v8f_zero();

  for (int kk = 0; kk < C1; kk += 32) {
    v16bf a;
    const int kb = kk + half * 16;
#pragma unroll
    for (int j = 0; j < 16; ++j) {
      a[j]     = (__bf16)abase[(size_t)(kb + j) * 16384];
    }
#pragma unroll
    for (int g = 0; g < 8; ++g) {
      const int n = n0 + g * 16 + l16;
      const __bf16* wrow = qwb + (size_t)n * C1 + kk + half * 16;
      v16bf bf;
#pragma unroll
      for (int j = 0; j < 16; ++j) bf[j] = wrow[j];
      acc[g] = wmma_bf16(a, bf, acc[g]);
    }
  }
  // bias
#pragma unroll
  for (int g = 0; g < 8; ++g) {
    const float bi = qb[n0 + g * 16 + l16];
#pragma unroll
    for (int r = 0; r < 8; ++r) acc[g][r] += bi;
  }
  // fused l2 normalization: each head spans two 16-wide accumulators
#pragma unroll
  for (int hp = 0; hp < 4; ++hp) {
    const int head = (n0 >> 5) + hp;
#pragma unroll
    for (int r = 0; r < 8; ++r) {
      float s = acc[2 * hp][r] * acc[2 * hp][r] +
                acc[2 * hp + 1][r] * acc[2 * hp + 1][r];
      s += __shfl_xor(s, 1); s += __shfl_xor(s, 2);
      s += __shfl_xor(s, 4); s += __shfl_xor(s, 8);   // reduce 16-lane half
      const float inv = 1.0f / fmaxf(sqrtf(s), 1e-12f);
      const int tok = mtile * 16 + r + half * 8;
      const int w2 = tok >> 8, tk = tok & 255;
      const size_t base = (((size_t)w2 * NHEAD + head) * L1 + tk) * HD;
      qn[base + l16]      = (__bf16)(acc[2 * hp][r] * inv);
      qn[base + 16 + l16] = (__bf16)(acc[2 * hp + 1][r] * inv);
    }
  }
}

// ---------------------------------------------------------------------------
// Kernel 3: kv = linear(window(mem_m)); k gets fused l2norm, v just bias
// A operand is the bf16 channels-last mem_m copy (contiguous in K).
// kn/vb layout: [win][head][tok(64)][c(32)]
// ---------------------------------------------------------------------------
__global__ __launch_bounds__(256) void k_kvnorm(
    const __bf16* __restrict__ mem_mb, const __bf16* __restrict__ kvwb,
    const float* __restrict__ kvb, __bf16* __restrict__ kn,
    __bf16* __restrict__ vb) {
  const int lane = threadIdx.x & 31, wv = threadIdx.x >> 5;
  const int half = lane >> 4, l16 = lane & 15;
  const int mtile = blockIdx.x * 8 + wv;
  const int n0 = blockIdx.y * 128;          // 0..384 over 512 kv channels

  const int t = mtile * 16 + l16;
  const int win = t >> 6, li = t & 63;
  const int ii = li >> 3, jj = li & 7;
  const int b = win >> 6, wq = win & 63;
  const int y = (wq >> 3) * 8 + ii, x = (wq & 7) * 8 + jj;
  const size_t tok2 = ((size_t)b * 64 + y) * 64 + x;
  const __bf16* abase = mem_mb + tok2 * C1;  // contiguous in k

  v8f acc[8];
#pragma unroll
  for (int g = 0; g < 8; ++g) acc[g] = v8f_zero();

  for (int kk = 0; kk < C1; kk += 32) {
    v16bf a;
    const int kb = kk + half * 16;
#pragma unroll
    for (int j = 0; j < 16; ++j) {
      a[j]     = abase[kb + j];
    }
#pragma unroll
    for (int g = 0; g < 8; ++g) {
      const int n = n0 + g * 16 + l16;
      const __bf16* wrow = kvwb + (size_t)n * C1 + kk + half * 16;
      v16bf bf;
#pragma unroll
      for (int j = 0; j < 16; ++j) bf[j] = wrow[j];
      acc[g] = wmma_bf16(a, bf, acc[g]);
    }
  }
#pragma unroll
  for (int g = 0; g < 8; ++g) {
    const float bi = kvb[n0 + g * 16 + l16];
#pragma unroll
    for (int r = 0; r < 8; ++r) acc[g][r] += bi;
  }

  if (n0 < C1) {  // ---- k half: l2norm then store bf16
#pragma unroll
    for (int hp = 0; hp < 4; ++hp) {
      const int head = (n0 >> 5) + hp;
#pragma unroll
      for (int r = 0; r < 8; ++r) {
        float s = acc[2 * hp][r] * acc[2 * hp][r] +
                  acc[2 * hp + 1][r] * acc[2 * hp + 1][r];
        s += __shfl_xor(s, 1); s += __shfl_xor(s, 2);
        s += __shfl_xor(s, 4); s += __shfl_xor(s, 8);
        const float inv = 1.0f / fmaxf(sqrtf(s), 1e-12f);
        const int tok = mtile * 16 + r + half * 8;
        const int w2 = tok >> 6, tk = tok & 63;
        const size_t base = (((size_t)w2 * NHEAD + head) * L2 + tk) * HD;
        kn[base + l16]      = (__bf16)(acc[2 * hp][r] * inv);
        kn[base + 16 + l16] = (__bf16)(acc[2 * hp + 1][r] * inv);
      }
    }
  } else {        // ---- v half: store bf16
    const int vn0 = n0 - C1;
#pragma unroll
    for (int hp = 0; hp < 4; ++hp) {
      const int head = (vn0 >> 5) + hp;
#pragma unroll
      for (int r = 0; r < 8; ++r) {
        const int tok = mtile * 16 + r + half * 8;
        const int w2 = tok >> 6, tk = tok & 63;
        const size_t base = (((size_t)w2 * NHEAD + head) * L2 + tk) * HD;
        vb[base + l16]      = (__bf16)acc[2 * hp][r];
        vb[base + 16 + l16] = (__bf16)acc[2 * hp + 1][r];
      }
    }
  }
}

// ---------------------------------------------------------------------------
// Kernel 4: fused attention per (win, head): QK^T -> softmax in registers
// -> bf16 probs via LDS (transpose to A-layout) -> @V -> bf16 out
// ---------------------------------------------------------------------------
__global__ __launch_bounds__(256) void k_attn(
    const __bf16* __restrict__ qn, const __bf16* __restrict__ kn,
    const __bf16* __restrict__ vb, const float* __restrict__ temperature,
    __bf16* __restrict__ aout) {
  __shared__ __bf16 ab[L1][L2];  // 32 KB: softmaxed probs, bf16

  const int win = blockIdx.x >> 3, head = blockIdx.x & 7;
  const int lane = threadIdx.x & 31, wv = threadIdx.x >> 5;
  const int half = lane >> 4, l16 = lane & 15;
  const float temp = temperature[head];

  const __bf16* qbase = qn + (((size_t)win * NHEAD + head) * L1) * HD;
  const __bf16* kbase = kn + (((size_t)win * NHEAD + head) * L2) * HD;
  const __bf16* vbase = vb + (((size_t)win * NHEAD + head) * L2) * HD;

  // ---- stage 1: scores + register softmax (each wave owns 2 M-tiles)
#pragma unroll
  for (int mth = 0; mth < 2; ++mth) {
    const int mt = wv * 2 + mth;
    v16bf a;
    const __bf16* arow = qbase + (size_t)(mt * 16 + l16) * HD + half * 16;
#pragma unroll
    for (int j = 0; j < 16; ++j) { a[j] = arow[j]; }

    v8f cc[4];
#pragma unroll
    for (int nt = 0; nt < 4; ++nt) {
      v16bf bf;
      const __bf16* brow = kbase + (size_t)(nt * 16 + l16) * HD + half * 16;
#pragma unroll
      for (int j = 0; j < 16; ++j) bf[j] = brow[j];
      cc[nt] = wmma_bf16(a, bf, v8f_zero());
    }
    // softmax over the 64 keys of each row: cross-lane within 16-lane half
#pragma unroll
    for (int r = 0; r < 8; ++r) {
      float v0 = cc[0][r] * temp, v1 = cc[1][r] * temp;
      float v2 = cc[2][r] * temp, v3 = cc[3][r] * temp;
      float mx = fmaxf(fmaxf(v0, v1), fmaxf(v2, v3));
      mx = fmaxf(mx, __shfl_xor(mx, 1)); mx = fmaxf(mx, __shfl_xor(mx, 2));
      mx = fmaxf(mx, __shfl_xor(mx, 4)); mx = fmaxf(mx, __shfl_xor(mx, 8));
      const float e0 = __expf(v0 - mx), e1 = __expf(v1 - mx);
      const float e2 = __expf(v2 - mx), e3 = __expf(v3 - mx);
      float sum = e0 + e1 + e2 + e3;
      sum += __shfl_xor(sum, 1); sum += __shfl_xor(sum, 2);
      sum += __shfl_xor(sum, 4); sum += __shfl_xor(sum, 8);
      const float inv = 1.0f / sum;
      const int row = mt * 16 + r + half * 8;
      ab[row][l16]      = (__bf16)(e0 * inv);
      ab[row][16 + l16] = (__bf16)(e1 * inv);
      ab[row][32 + l16] = (__bf16)(e2 * inv);
      ab[row][48 + l16] = (__bf16)(e3 * inv);
    }
  }
  __syncthreads();

  // ---- stage 2: out = probs(256x64) @ V(64x32)
#pragma unroll
  for (int mth = 0; mth < 2; ++mth) {
    const int mt = wv * 2 + mth;
    v8f c0 = v8f_zero(), c1 = v8f_zero();
    for (int kk = 0; kk < L2; kk += 32) {
      v16bf a;
      const int kb = kk + half * 16;
      const __bf16* arow = &ab[mt * 16 + l16][0];
#pragma unroll
      for (int j = 0; j < 16; ++j) { a[j] = arow[kb + j]; }
#pragma unroll
      for (int nt = 0; nt < 2; ++nt) {
        v16bf bf;
        const __bf16* bcol = vbase + (size_t)(kk + half * 16) * HD + nt * 16 + l16;
#pragma unroll
        for (int j = 0; j < 16; ++j) bf[j] = bcol[(size_t)j * HD];
        if (nt == 0) c0 = wmma_bf16(a, bf, c0);
        else         c1 = wmma_bf16(a, bf, c1);
      }
    }
#pragma unroll
    for (int r = 0; r < 8; ++r) {
      const int m = mt * 16 + r + half * 8;
      const size_t base = ((size_t)win * L1 + m) * C1 + head * HD;
      aout[base + l16]      = (__bf16)c0[r];
      aout[base + 16 + l16] = (__bf16)c1[r];
    }
  }
}

// ---------------------------------------------------------------------------
// Kernel 5: output projection (M=131072, N=256, K=256) + window_reverse
// ---------------------------------------------------------------------------
__global__ __launch_bounds__(256) void k_proj(
    const __bf16* __restrict__ ain, const __bf16* __restrict__ pwb16,
    const float* __restrict__ pb, float* __restrict__ out) {
  const int lane = threadIdx.x & 31, wv = threadIdx.x >> 5;
  const int half = lane >> 4, l16 = lane & 15;
  const int mtile = blockIdx.x * 8 + wv;
  const int n0 = blockIdx.y * 128;

  const __bf16* arow0 = ain + (size_t)(mtile * 16 + l16) * C1;

  v8f acc[8];
#pragma unroll
  for (int g = 0; g < 8; ++g) acc[g] = v8f_zero();

  for (int kk = 0; kk < C1; kk += 32) {
    v16bf a;
    const int kb = kk + half * 16;
#pragma unroll
    for (int j = 0; j < 16; ++j) { a[j] = arow0[kb + j]; }
#pragma unroll
    for (int g = 0; g < 8; ++g) {
      const int n = n0 + g * 16 + l16;
      const __bf16* wrow = pwb16 + (size_t)n * C1 + kk + half * 16;
      v16bf bf;
#pragma unroll
      for (int j = 0; j < 16; ++j) bf[j] = wrow[j];
      acc[g] = wmma_bf16(a, bf, acc[g]);
    }
  }
#pragma unroll
  for (int g = 0; g < 8; ++g) {
    const int n = n0 + g * 16 + l16;
    const float bi = pb[n];
#pragma unroll
    for (int r = 0; r < 8; ++r) {
      const int tok = mtile * 16 + r + half * 8;
      const int win = tok >> 8, li = tok & 255;
      const int ii = li >> 4, jj = li & 15;
      const int b = win >> 6, wq = win & 63;
      const int y = (wq >> 3) * 16 + ii, x = (wq & 7) * 16 + jj;
      out[(((size_t)b * C1 + n) * H1c + y) * W1c + x] = acc[g][r] + bi;
    }
  }
}

// ---------------------------------------------------------------------------
// Kernel 6: ghost branch (elementwise), adds into out. mem_m is L2-resident.
// ---------------------------------------------------------------------------
__device__ __forceinline__ float bilin_mem(const float* __restrict__ mm,
                                           int b, int c, int y, int x) {
  const float fy = (y + 0.5f) * 0.5f - 0.5f;
  const float fx = (x + 0.5f) * 0.5f - 0.5f;
  const int y0 = (int)floorf(fy), x0 = (int)floorf(fx);
  const float wy = fy - (float)y0, wx = fx - (float)x0;
  const int y0c = min(max(y0, 0), 63), y1c = min(max(y0 + 1, 0), 63);
  const int x0c = min(max(x0, 0), 63), x1c = min(max(x0 + 1, 0), 63);
  const float* base = mm + (size_t)b * 4096 * C1 + c;
  const float v00 = base[((size_t)y0c * 64 + x0c) * C1];
  const float v01 = base[((size_t)y0c * 64 + x1c) * C1];
  const float v10 = base[((size_t)y1c * 64 + x0c) * C1];
  const float v11 = base[((size_t)y1c * 64 + x1c) * C1];
  return v00 * (1.f - wy) * (1.f - wx) + v01 * (1.f - wy) * wx +
         v10 * wy * (1.f - wx) + v11 * wy * wx;
}

__global__ __launch_bounds__(256) void k_ghost(
    const float* __restrict__ tgt, const float* __restrict__ mem_m,
    const float* __restrict__ dww, const float* __restrict__ bng,
    const float* __restrict__ bnb, const float* __restrict__ pww,
    const float* __restrict__ pwb, float* __restrict__ out) {
  const size_t idx = (size_t)blockIdx.x * 256 + threadIdx.x;
  const int x = (int)(idx & 127);
  const int y = (int)((idx >> 7) & 127);
  const int c = (int)((idx >> 14) & 255);
  const int b = (int)(idx >> 22);

  const float* tch = tgt + ((size_t)b * C1 + c) * 16384;
  const float* w1 = dww + (size_t)c * 9;
  const float* w2 = dww + (size_t)(C1 + c) * 9;

  float g1 = 0.f, g2 = 0.f;
#pragma unroll
  for (int dy = -1; dy <= 1; ++dy) {
#pragma unroll
    for (int dx = -1; dx <= 1; ++dx) {
      const int yy = y + dy, xx = x + dx;
      float tv = 0.f, mv = 0.f;
      if (yy >= 0 && yy < 128 && xx >= 0 && xx < 128) {
        tv = tch[(size_t)yy * 128 + xx];
        mv = bilin_mem(mem_m, b, c, yy, xx);
      }
      const int wi = (dy + 1) * 3 + (dx + 1);
      g1 += w1[wi] * tv;
      g2 += w2[wi] * mv;
    }
  }
  float g = g1 * g2;
  g = g * (bng[c] * 0.99999500003f) + bnb[c];  // 1/sqrt(1+1e-5)
  g = (g >= 0.f) ? g : 0.2f * g;
  g = g * pww[c] + pwb[c];
  out[idx] += g;
}

// ---------------------------------------------------------------------------
extern "C" void kernel_launch(void* const* d_in, const int* in_sizes, int n_in,
                              void* d_out, int out_size, void* d_ws,
                              size_t ws_size, hipStream_t stream) {
  const float* tgt  = (const float*)d_in[0];
  const float* mem  = (const float*)d_in[1];
  const float* mcw  = (const float*)d_in[2];
  const float* mcb  = (const float*)d_in[3];
  const float* qw   = (const float*)d_in[4];
  const float* qb   = (const float*)d_in[5];
  const float* kvw  = (const float*)d_in[6];
  const float* kvb  = (const float*)d_in[7];
  const float* pw   = (const float*)d_in[8];
  const float* pb   = (const float*)d_in[9];
  const float* temp = (const float*)d_in[10];
  const float* dww  = (const float*)d_in[11];
  const float* bng  = (const float*)d_in[12];
  const float* bnb  = (const float*)d_in[13];
  const float* pww  = (const float*)d_in[14];
  const float* pwb  = (const float*)d_in[15];
  float* out = (float*)d_out;

  char* ws = (char*)d_ws;
  float*  mem_m  = (float*)(ws);                   //  33.6 MB f32 chan-last
  __bf16* mem_mb = (__bf16*)(ws + 33554432);       //  16.8 MB bf16 chan-last
  __bf16* qn     = (__bf16*)(ws + 50331648);       //  67.1 MB
  __bf16* kn     = (__bf16*)(ws + 117440512);      //  16.8 MB
  __bf16* vbuf   = (__bf16*)(ws + 134217728);      //  16.8 MB
  __bf16* aout   = (__bf16*)(ws + 150994944);      //  67.1 MB
  __bf16* wb_mc  = (__bf16*)(ws + 218103808);      //  64 KB
  __bf16* wb_q   = (__bf16*)(ws + 218169344);      // 128 KB
  __bf16* wb_kv  = (__bf16*)(ws + 218300416);      // 256 KB
  __bf16* wb_pj  = (__bf16*)(ws + 218562560);      // 128 KB

  // weight pre-conversion (tiny)
  k_cvt<<<128, 256, 0, stream>>>(mcw, wb_mc, 256 * 128);
  k_cvt<<<256, 256, 0, stream>>>(qw,  wb_q,  256 * 256);
  k_cvt<<<512, 256, 0, stream>>>(kvw, wb_kv, 512 * 256);
  k_cvt<<<256, 256, 0, stream>>>(pw,  wb_pj, 256 * 256);

  k_conv1x1<<<dim3(256, 2),  256, 0, stream>>>(mem, wb_mc, mcb, mem_m, mem_mb);
  k_qnorm  <<<dim3(1024, 2), 256, 0, stream>>>(tgt, wb_q, qb, qn);
  k_kvnorm <<<dim3(256, 4),  256, 0, stream>>>(mem_mb, wb_kv, kvb, kn, vbuf);
  k_attn   <<<4096,          256, 0, stream>>>(qn, kn, vbuf, temp, aout);
  k_proj   <<<dim3(1024, 2), 256, 0, stream>>>(aout, wb_pj, pb, out);
  k_ghost  <<<131072,        256, 0, stream>>>(tgt, mem_m, dww, bng, bnb,
                                               pww, pwb, out);
}
